// VLAD_41592463294661
// MI455X (gfx1250) — matrix operand order repaired and checked by reading, not compiled, not run on hardware
//
#include <hip/hip_runtime.h>

// ---------------------------------------------------------------------------
// NetVLAD head for MI455X (gfx1250): bf16 WMMA (16x16x32, f32 accum) pipeline
// Round 5 (final polish): prefetch with max locality so global_prefetch_b8
// pulls the next input tile into WGP-level cache (re-read by 2 waves/WGP),
// instead of GL2-only (SCOPE_SYS) as emitted with locality 0.
// ---------------------------------------------------------------------------

typedef __bf16 bf16_t;
typedef __attribute__((ext_vector_type(8)))  bf16_t v8bf;
typedef __attribute__((ext_vector_type(16))) bf16_t v16bf;
typedef __attribute__((ext_vector_type(8)))  float  v8f;

#define NIMG   64
#define PIX    3136      // 56*56 pixels per image
#define CIN    512
#define DD     128
#define KC     64        // num centers
#define CHUNKS 7         // workgroups per image
#define TILES  7         // 64-pixel row tiles per chunk  (7*7*64 = 3136)

// ---- LDS layout (bytes); all offsets multiple of 16 ----
#define SM_WT    0u        // bf16 [128][520]  w_transform^T      133120 B
#define SM_WA    133120u   // bf16 [64][136]   w_assign^T          17408 B
#define SM_FR    150528u   // bf16 [64][136]   f row-major         17408 B
#define SM_FT    167936u   // bf16 [128][72]   f transposed        18432 B
#define SM_ATR   186368u   // bf16 [64][72]    a transposed         9216 B
#define SM_LG    195584u   // f32  [64][68]    logits              17408 B
#define SM_ASUM  212992u   // f32  [64]                               256 B
#define SM_B     213248u   // f32  [64]        bias                  256 B
#define SM_TOTAL 213504u

// padded row strides (elements)
#define WT_S 520
#define WA_S 136
#define FR_S 136
#define FT_S 72
#define ATR_S 72
#define LG_S 68

// workspace layout (bytes)
#define WS_WT    0u                      // bf16 [128][512]
#define WS_WA    131072u                 // bf16 [64][128]
#define WS_AFP   147456u                 // f32  [NIMG*CHUNKS][64][128]
#define WS_ASUMP (147456u + 14680064u)   // f32  [NIMG*CHUNKS][64]

// Two contiguous 8-element (16B, 16B-aligned) bf16 chunks -> one fragment.
static __device__ __forceinline__ v16bf frag2(const bf16_t* p0, const bf16_t* p1) {
  v8bf a = *(const v8bf*)p0;
  v8bf b = *(const v8bf*)p1;
  return __builtin_shufflevector(a, b, 0,1,2,3,4,5,6,7,8,9,10,11,12,13,14,15);
}

// A-fragment straight from global f32 (row-major, per-lane row pointer p
// already includes +16h): chunk0 = p[0..7] (K=kb+16h+e), chunk1 = p[8..15].
static __device__ __forceinline__ v16bf fragA_g(const float* __restrict__ p) {
  float4 a = *(const float4*)(p);
  float4 b = *(const float4*)(p + 4);
  float4 c = *(const float4*)(p + 8);
  float4 d = *(const float4*)(p + 12);
  v16bf r;
  r[0]  = (bf16_t)a.x; r[1]  = (bf16_t)a.y; r[2]  = (bf16_t)a.z; r[3]  = (bf16_t)a.w;
  r[4]  = (bf16_t)b.x; r[5]  = (bf16_t)b.y; r[6]  = (bf16_t)b.z; r[7]  = (bf16_t)b.w;
  r[8]  = (bf16_t)c.x; r[9]  = (bf16_t)c.y; r[10] = (bf16_t)c.z; r[11] = (bf16_t)c.w;
  r[12] = (bf16_t)d.x; r[13] = (bf16_t)d.y; r[14] = (bf16_t)d.z; r[15] = (bf16_t)d.w;
  return r;
}

static __device__ __forceinline__ v8f wmma_bf16(v16bf A, v16bf B, v8f C) {
  return __builtin_amdgcn_wmma_f32_16x16x32_bf16(
      /*neg_a=*/false, A, /*neg_b=*/false, B,
      /*c_mod=*/(short)0, C, /*reuse_a=*/false, /*reuse_b=*/false);
}

// ---------------------------------------------------------------------------
// prep: f32 weights -> bf16, transposed so B-fragments are contiguous in LDS
// ---------------------------------------------------------------------------
extern "C" __global__ void vlad_prep(const float* __restrict__ wt,  // [512][128]
                                     const float* __restrict__ wa,  // [128][64]
                                     bf16_t* __restrict__ wT_t,     // [128][512]
                                     bf16_t* __restrict__ wA_t) {   // [64][128]
  int i = blockIdx.x * 256 + threadIdx.x;
  if (i < CIN * DD) {
    int d = i >> 9;          // / 512
    int c = i & 511;
    wT_t[i] = (bf16_t)wt[c * DD + d];
  }
  int j = i - CIN * DD;
  if (j >= 0 && j < KC * DD) {
    int k = j >> 7;
    int d = j & 127;
    wA_t[j] = (bf16_t)wa[d * KC + k];
  }
}

// ---------------------------------------------------------------------------
// stage1: per (image, chunk): GEMM1 -> GEMM2 -> softmax -> GEMM3 partials
// ---------------------------------------------------------------------------
extern "C" __global__ void __launch_bounds__(256, 1)
vlad_stage1(const float* __restrict__ inputs,     // [NIMG*PIX][512]
            const bf16_t* __restrict__ wT_t,      // [128][512]
            const bf16_t* __restrict__ wA_t,      // [64][128]
            const float* __restrict__ b_assign,   // [64]
            float* __restrict__ afp,              // [NIMG*CHUNKS][64][128]
            float* __restrict__ asump)            // [NIMG*CHUNKS][64]
{
  extern __shared__ char smem[];
  bf16_t* WTm  = (bf16_t*)(smem + SM_WT);
  bf16_t* WAm  = (bf16_t*)(smem + SM_WA);
  bf16_t* FRm  = (bf16_t*)(smem + SM_FR);
  bf16_t* FTm  = (bf16_t*)(smem + SM_FT);
  bf16_t* ATRm = (bf16_t*)(smem + SM_ATR);
  float*  LGm  = (float*)(smem + SM_LG);
  float*  ASUM = (float*)(smem + SM_ASUM);
  float*  Bb   = (float*)(smem + SM_B);

  const int tid  = threadIdx.x;
  const int lane = tid & 31;
  const int w    = tid >> 5;      // wave id 0..7
  const int ln   = lane & 15;
  const int h    = lane >> 4;
  const int img   = blockIdx.y;
  const int chunk = blockIdx.x;

  // ---- phase 0: stage weights into LDS (padded rows) ----
  for (int v = tid; v < (DD * CIN) / 8; v += 256) {     // 8192 vec8
    int n = v >> 6;                 // row (d), 64 vec8 per row
    int k = (v & 63) << 3;
    uint4 dta = ((const uint4*)wT_t)[v];
    *(uint4*)(WTm + n * WT_S + k) = dta;
  }
  for (int v = tid; v < (KC * DD) / 8; v += 256) {      // 1024 vec8
    int n = v >> 4;
    int k = (v & 15) << 3;
    uint4 dta = ((const uint4*)wA_t)[v];
    *(uint4*)(WAm + n * WA_S + k) = dta;
  }
  if (tid < KC) { Bb[tid] = b_assign[tid]; ASUM[tid] = 0.f; }
  __syncthreads();   // WT/WA visible before first GEMM1

  // output tiling: 8 waves cover a 64x128 block as 4 row-strips x 2 col-halves
  const int rs  = (w >> 1) * 16;   // 16-row strip
  const int cs  = (w & 1) * 64;    // 64-col half (4 N-tiles)
  const int rs2 = (w >> 1) * 16;   // GEMM2: 64x64 -> 2 N-tiles per wave
  const int cs2 = (w & 1) * 32;

  // per-wave fragment base pointers (constant across tiles)
  const bf16_t* brow0 = WTm + (cs +  0 + ln) * WT_S + h * 16;
  const bf16_t* brow1 = WTm + (cs + 16 + ln) * WT_S + h * 16;
  const bf16_t* brow2 = WTm + (cs + 32 + ln) * WT_S + h * 16;
  const bf16_t* brow3 = WTm + (cs + 48 + ln) * WT_S + h * 16;
  const bf16_t* a2row = FRm + (rs2 + ln) * FR_S + h * 16;    // GEMM2 A base
  const bf16_t* w2r0  = WAm + (cs2 +  0 + ln) * WA_S + h * 16;
  const bf16_t* w2r1  = WAm + (cs2 + 16 + ln) * WA_S + h * 16;
  const bf16_t* a3row = ATRm + (rs + ln) * ATR_S + h * 16;   // GEMM3 A base
  const bf16_t* f3r0  = FTm + (cs +  0 + ln) * FT_S + h * 16;
  const bf16_t* f3r1  = FTm + (cs + 16 + ln) * FT_S + h * 16;
  const bf16_t* f3r2  = FTm + (cs + 32 + ln) * FT_S + h * 16;
  const bf16_t* f3r3  = FTm + (cs + 48 + ln) * FT_S + h * 16;

  const v8f z = {0.f,0.f,0.f,0.f,0.f,0.f,0.f,0.f};
  v8f afacc[4];                     // persistent af accumulator (WMMA C chain)
  afacc[0] = z; afacc[1] = z; afacc[2] = z; afacc[3] = z;

  for (int t = 0; t < TILES; ++t) {
    const long rowbase = (long)img * PIX + (long)(chunk * TILES + t) * 64;

    // prefetch next tile into near caches (gfx1250 global_prefetch_b8);
    // locality 3 -> WGP-scope prefetch: rows are re-read by both column-half
    // waves of this WGP, so WGP$ residency pays off twice.
    if (t + 1 < TILES) {
      const float* nsrc = inputs + (rowbase + 64) * CIN;
#pragma unroll
      for (int q = 0; q < 4; ++q)
        __builtin_prefetch(nsrc + (q * 256 + tid) * 32, 0, 3);
    }

    // ---- GEMM1: f[64x128] = in[64x512] * wT[512x128] ----
    // A from global f32 (VMEM/LOADcnt) with in-register bf16 convert;
    // B from LDS (DScnt); per k-step all loads precede the 4 WMMAs.
    const float* grow = inputs + (rowbase + rs + ln) * (long)CIN + h * 16;
    v8f f1acc[4];
    f1acc[0] = z; f1acc[1] = z; f1acc[2] = z; f1acc[3] = z;
#pragma unroll 4
    for (int kk = 0; kk < 16; ++kk) {
      const int kb = kk * 32;
      v16bf A  = fragA_g(grow + kb);
      v16bf B0 = frag2(brow0 + kb, brow0 + kb + 8);
      v16bf B1 = frag2(brow1 + kb, brow1 + kb + 8);
      v16bf B2 = frag2(brow2 + kb, brow2 + kb + 8);
      v16bf B3 = frag2(brow3 + kb, brow3 + kb + 8);
      f1acc[0] = wmma_bf16(A, B0, f1acc[0]);
      f1acc[1] = wmma_bf16(A, B1, f1acc[1]);
      f1acc[2] = wmma_bf16(A, B2, f1acc[2]);
      f1acc[3] = wmma_bf16(A, B3, f1acc[3]);
    }
    __syncthreads();   // previous tile's GEMM2/GEMM3 finished with FR/FT/ATR

    // write f (bf16) row-major + transposed; C layout: m = rs + r + 8h
#pragma unroll
    for (int j = 0; j < 4; ++j)
#pragma unroll
      for (int r = 0; r < 8; ++r) {
        int m = rs + r + 8 * h;
        int n = cs + j * 16 + ln;
        bf16_t bv = (bf16_t)f1acc[j][r];
        FRm[m * FR_S + n] = bv;
        FTm[n * FT_S + m] = bv;
      }
    __syncthreads();

    // ---- GEMM2: logits[64x64] = f[64x128] * wA[128x64] ----
    v8f lacc[2];
    lacc[0] = z; lacc[1] = z;
#pragma unroll
    for (int kk = 0; kk < 4; ++kk) {
      const int kb = kk * 32;
      v16bf A  = frag2(a2row + kb, a2row + kb + 8);
      v16bf B0 = frag2(w2r0 + kb,  w2r0 + kb + 8);
      v16bf B1 = frag2(w2r1 + kb,  w2r1 + kb + 8);
      lacc[0] = wmma_bf16(A, B0, lacc[0]);
      lacc[1] = wmma_bf16(A, B1, lacc[1]);
    }
#pragma unroll
    for (int j = 0; j < 2; ++j)
#pragma unroll
      for (int r = 0; r < 8; ++r) {
        int m = rs2 + r + 8 * h;
        int n = cs2 + j * 16 + ln;
        LGm[m * LG_S + n] = lacc[j][r] + Bb[n];
      }
    __syncthreads();

    // ---- softmax per pixel row (f32), write a^T in bf16 ----
    if (tid < 64) {
      float* lg = LGm + tid * LG_S;
      float mx = lg[0];
      for (int k = 1; k < KC; ++k) mx = fmaxf(mx, lg[k]);
      float s = 0.f;
      for (int k = 0; k < KC; ++k) { float e = __expf(lg[k] - mx); s += e; lg[k] = e; }
      float inv = 1.0f / s;
      for (int k = 0; k < KC; ++k) ATRm[k * ATR_S + tid] = (bf16_t)(lg[k] * inv);
    }
    __syncthreads();

    // a_sum[k] accumulation (thread k owns ASUM[k]; consistent with bf16 a)
    if (tid < KC) {
      float s = 0.f;
      for (int m = 0; m < 64; ++m) s += (float)ATRm[tid * ATR_S + m];
      ASUM[tid] += s;
    }

    // ---- GEMM3: af[64x128] += a^T[64 x 64pix] * f[64pix x 128] ----
    // accumulate directly into the persistent C registers (no LDS RMW)
#pragma unroll
    for (int pp = 0; pp < 2; ++pp) {
      const int pb = pp * 32;
      v16bf A  = frag2(a3row + pb, a3row + pb + 8);
      v16bf B0 = frag2(f3r0 + pb,  f3r0 + pb + 8);
      v16bf B1 = frag2(f3r1 + pb,  f3r1 + pb + 8);
      v16bf B2 = frag2(f3r2 + pb,  f3r2 + pb + 8);
      v16bf B3 = frag2(f3r3 + pb,  f3r3 + pb + 8);
      afacc[0] = wmma_bf16(A, B0, afacc[0]);
      afacc[1] = wmma_bf16(A, B1, afacc[1]);
      afacc[2] = wmma_bf16(A, B2, afacc[2]);
      afacc[3] = wmma_bf16(A, B3, afacc[3]);
    }
  }

  // ---- epilogue: write af / a_sum partials ----
  const int pc = img * CHUNKS + chunk;
#pragma unroll
  for (int j = 0; j < 4; ++j)
#pragma unroll
    for (int r = 0; r < 8; ++r) {
      int k = rs + r + 8 * h;
      int d = cs + j * 16 + ln;
      afp[(long)(pc * KC + k) * DD + d] = afacc[j][r];
    }
  if (tid < KC) asump[pc * KC + tid] = ASUM[tid];
}

// ---------------------------------------------------------------------------
// finalize: reduce partials, v = af - a_sum*centers, intra + global L2 norm
// ---------------------------------------------------------------------------
extern "C" __global__ void __launch_bounds__(256)
vlad_finalize(const float* __restrict__ afp,
              const float* __restrict__ asump,
              const float* __restrict__ centers,   // [64][128]
              float* __restrict__ out)             // [NIMG][8192]
{
  __shared__ float asumL[64];
  __shared__ float red[256];
  __shared__ float scaleL[64];
  __shared__ float nss[64];
  __shared__ float gscale;

  const int img = blockIdx.x;
  const int t = threadIdx.x;
  const int k = t >> 2;            // 4 threads per center row
  const int dq = (t & 3) * 32;

  if (t < 64) {
    float s = 0.f;
    for (int c = 0; c < CHUNKS; ++c) s += asump[(img * CHUNKS + c) * KC + t];
    asumL[t] = s;
  }
  __syncthreads();

  float v[32];
  float ss = 0.f;
  const long base = (long)img * CHUNKS * (KC * DD);
  const float ak = asumL[k];
#pragma unroll 4
  for (int i = 0; i < 32; ++i) {
    const int e = k * DD + dq + i;
    float s = 0.f;
    for (int c = 0; c < CHUNKS; ++c) s += afp[base + (long)c * (KC * DD) + e];
    s -= ak * centers[e];
    v[i] = s;
    ss += s * s;
  }
  red[t] = ss;
  __syncthreads();
  if ((t & 3) == 0) {
    float tot = red[t] + red[t + 1] + red[t + 2] + red[t + 3];
    float sc = rsqrtf(fmaxf(tot, 1e-12f));
    scaleL[k] = sc;
    nss[k] = tot * sc * sc;        // squared norm after intra-normalization
  }
  __syncthreads();
  if (t == 0) {
    float g = 0.f;
    for (int q = 0; q < 64; ++q) g += nss[q];
    gscale = rsqrtf(fmaxf(g, 1e-12f));
  }
  __syncthreads();
  const float sc = scaleL[k] * gscale;
#pragma unroll 4
  for (int i = 0; i < 32; ++i)
    out[(long)img * (KC * DD) + k * DD + dq + i] = v[i] * sc;
}

// ---------------------------------------------------------------------------
extern "C" void kernel_launch(void* const* d_in, const int* in_sizes, int n_in,
                              void* d_out, int out_size, void* d_ws, size_t ws_size,
                              hipStream_t stream) {
  (void)in_sizes; (void)n_in; (void)out_size; (void)ws_size;
  const float* inputs      = (const float*)d_in[0];
  const float* w_transform = (const float*)d_in[1];
  const float* w_assign    = (const float*)d_in[2];
  const float* b_assign    = (const float*)d_in[3];
  const float* centers     = (const float*)d_in[4];
  float* out = (float*)d_out;

  char* ws = (char*)d_ws;
  bf16_t* wT_t  = (bf16_t*)(ws + WS_WT);
  bf16_t* wA_t  = (bf16_t*)(ws + WS_WA);
  float*  afp   = (float*)(ws + WS_AFP);
  float*  asump = (float*)(ws + WS_ASUMP);

  // allow ~214KB dynamic LDS (CDNA5 WGP: up to 320KB per workgroup)
  hipFuncSetAttribute(reinterpret_cast<const void*>(vlad_stage1),
                      hipFuncAttributeMaxDynamicSharedMemorySize, (int)SM_TOTAL);

  vlad_prep<<<(CIN * DD + KC * DD + 255) / 256, 256, 0, stream>>>(
      w_transform, w_assign, wT_t, wA_t);
  vlad_stage1<<<dim3(CHUNKS, NIMG), 256, SM_TOTAL, stream>>>(
      inputs, wT_t, wA_t, b_assign, afp, asump);
  vlad_finalize<<<NIMG, 256, 0, stream>>>(afp, asump, centers, out);
}
